// RelativeGridAttention_68539088109939
// MI455X (gfx1250) — matrix operand order repaired and checked by reading, not compiled, not run on hardware
//
#include <hip/hip_runtime.h>

// ---------------------------------------------------------------------------
// Fused RelativeGridAttention for MI455X (gfx1250, wave32, WMMA bf16)
//   scores = Q.K^T / 16  ->  + rel_bias gather  ->  softmax(1024)
//   -> per-map 3x3 conv (1->8, ReLU) -> 3x3 conv (8->1) -> residual add
// One block per (batch, 16-query tile): 1024 blocks x 256 threads (8 waves).
// Score tile (16x1024 f32) held entirely in wave registers (8 v8f accums/wave).
// ---------------------------------------------------------------------------

typedef __attribute__((ext_vector_type(8)))  float   v8f;
typedef __attribute__((ext_vector_type(8)))  __bf16  v8bf;
typedef __attribute__((ext_vector_type(16))) __bf16  v16bf;

#define NB   16
#define NQ   1024
#define NC   256
#define NH   32
#define NW   32
#define NHW  1024

// --- LDS arena (phase-reused), byte offsets --------------------------------
// phase 1 (gemm + bias + softmax):
#define OFF_A        0       // 16*256 bf16           = 8192
#define OFF_RELBIAS  8192    // 63*63 f32             = 15876 -> 24068
#define OFF_POS      24080   // 32 i32                -> 24208
#define OFF_RED      24208   // 16*128 f32            -> 32400
// phase 2 (convs), reuses phase-1 space:
#define OFF_IMG      0       // 32*32 f32             = 4096
#define OFF_H1       4096    // 8*34*34 f32           = 36992 -> 41088
#define OFF_CW       41088   // 153 f32               -> 41700
// survives both phases (outside conv region):
#define OFF_ROWMAX   41728   // 16 f32
#define OFF_ROWSUM   41792   // 16 f32 (reciprocal sums)
#define SMEM_BYTES   41856

__device__ __forceinline__ float v8_get(v8f v, int i) {
  switch (i & 7) {
    case 0: return v[0]; case 1: return v[1];
    case 2: return v[2]; case 3: return v[3];
    case 4: return v[4]; case 5: return v[5];
    case 6: return v[6]; default: return v[7];
  }
}

__global__ __launch_bounds__(256)
void fused_grid_attn_conv(const float* __restrict__ queries,
                          const float* __restrict__ keys,
                          const int*   __restrict__ pos,
                          const float* __restrict__ rel_bias,
                          const float* __restrict__ c1w,
                          const float* __restrict__ c1b,
                          const float* __restrict__ c2w,
                          const float* __restrict__ c2b,
                          float*       __restrict__ out) {
  __shared__ __align__(16) unsigned char smem[SMEM_BYTES];
  __bf16* sA       = (__bf16*)(smem + OFF_A);
  float*  sRelBias = (float*)(smem + OFF_RELBIAS);
  int*    sPos     = (int*)(smem + OFF_POS);
  float*  sRed     = (float*)(smem + OFF_RED);
  float*  sImg     = (float*)(smem + OFF_IMG);
  float*  sH1      = (float*)(smem + OFF_H1);
  float*  sCW      = (float*)(smem + OFF_CW);
  float*  sRowMax  = (float*)(smem + OFF_ROWMAX);
  float*  sRowSum  = (float*)(smem + OFF_ROWSUM);

  const int tid    = threadIdx.x;
  const int wv     = tid >> 5;          // wave 0..7
  const int lane   = tid & 31;
  const int lane15 = lane & 15;
  const int hi     = lane >> 4;         // which 16-lane half

  const int b  = blockIdx.x >> 6;       // batch
  const int qt = blockIdx.x & 63;       // 16-query tile

  // ---- stage A tile (queries, f32->bf16) + rel_bias + pos into LDS --------
  {
    const int row = tid >> 4;                 // 0..15
    const int col = (tid & 15) << 4;          // 0..240 step 16
    const float4* qp = (const float4*)(queries +
        ((size_t)(b * NQ + qt * 16 + row)) * NC + col);
    float4 f0 = qp[0], f1 = qp[1], f2 = qp[2], f3 = qp[3];
    v8bf lo, hi8;
    lo[0]=(__bf16)f0.x; lo[1]=(__bf16)f0.y; lo[2]=(__bf16)f0.z; lo[3]=(__bf16)f0.w;
    lo[4]=(__bf16)f1.x; lo[5]=(__bf16)f1.y; lo[6]=(__bf16)f1.z; lo[7]=(__bf16)f1.w;
    hi8[0]=(__bf16)f2.x; hi8[1]=(__bf16)f2.y; hi8[2]=(__bf16)f2.z; hi8[3]=(__bf16)f2.w;
    hi8[4]=(__bf16)f3.x; hi8[5]=(__bf16)f3.y; hi8[6]=(__bf16)f3.z; hi8[7]=(__bf16)f3.w;
    *(v8bf*)(sA + row * NC + col)     = lo;
    *(v8bf*)(sA + row * NC + col + 8) = hi8;
  }
  for (int i = tid; i < 63 * 63; i += 256) sRelBias[i] = rel_bias[i];
  if (tid < 32) sPos[tid] = pos[(size_t)(b * NQ + qt * 16) * 2 + tid];
  __syncthreads();

  // ---- GEMM: 16x1024 scores, bf16 WMMA, f32 accumulate --------------------
  v8f acc[8];
  {
    v8f zero = {0.f, 0.f, 0.f, 0.f, 0.f, 0.f, 0.f, 0.f};
#pragma unroll
    for (int t = 0; t < 8; ++t) acc[t] = zero;
  }

#pragma unroll
  for (int kk = 0; kk < 8; ++kk) {          // K chunks of 32 channels
    // A fragment (16x32 bf16) from LDS; layout: K = hi*16 + {0..15}
    const __bf16* ap = sA + (lane15 << 8) + (kk << 5) + (hi << 4);
    v8bf alo  = *(const v8bf*)ap;
    v8bf ahi8 = *(const v8bf*)(ap + 8);
    v16bf afrag = __builtin_shufflevector(alo, ahi8,
        0,1,2,3,4,5,6,7,8,9,10,11,12,13,14,15);

#pragma unroll
    for (int t = 0; t < 8; ++t) {
      const int n = ((wv * 8 + t) << 4) + lane15;       // key position 0..1023
      const float* kp = keys + ((size_t)(b * NHW + n)) * NC + (kk << 5) + (hi << 4);
      if (kk < 7) __builtin_prefetch((const void*)(kp + 32), 0, 1);
      const float4* kp4 = (const float4*)kp;            // 64B aligned
      float4 f0 = kp4[0], f1 = kp4[1], f2 = kp4[2], f3 = kp4[3];
      v16bf bfrag;                                      // B: K = hi*16 + i
      bfrag[0]=(__bf16)f0.x;  bfrag[1]=(__bf16)f0.y;  bfrag[2]=(__bf16)f0.z;  bfrag[3]=(__bf16)f0.w;
      bfrag[4]=(__bf16)f1.x;  bfrag[5]=(__bf16)f1.y;  bfrag[6]=(__bf16)f1.z;  bfrag[7]=(__bf16)f1.w;
      bfrag[8]=(__bf16)f2.x;  bfrag[9]=(__bf16)f2.y;  bfrag[10]=(__bf16)f2.z; bfrag[11]=(__bf16)f2.w;
      bfrag[12]=(__bf16)f3.x; bfrag[13]=(__bf16)f3.y; bfrag[14]=(__bf16)f3.z; bfrag[15]=(__bf16)f3.w;
      acc[t] = __builtin_amdgcn_wmma_f32_16x16x32_bf16(
          false, afrag, false, bfrag, (short)0, acc[t], false, false);
    }
  }

  // ---- bias add (rel_bias gather) + row max partials ----------------------
#pragma unroll
  for (int vr = 0; vr < 8; ++vr) {
    const int M  = vr + 8 * hi;
    const int p0 = sPos[2 * M];
    const int p1 = sPos[2 * M + 1];
    float m = -3.4e38f;
#pragma unroll
    for (int t = 0; t < 8; ++t) {
      const int N  = ((wv * 8 + t) << 4) + lane15;
      const int hc = N >> 5, wc = N & 31;
      const float bias = sRelBias[(hc - p0 + 31) * 63 + (wc - p1 + 31)];
      const float logit = acc[t][vr] * 0.0625f + bias;   // 1/sqrt(256)
      acc[t][vr] = logit;
      m = fmaxf(m, logit);
    }
    sRed[M * 128 + wv * 16 + lane15] = m;
  }
  __syncthreads();
  if (tid < 16) {
    float m = -3.4e38f;
    const float* r0 = sRed + tid * 128;
    for (int j = 0; j < 128; ++j) m = fmaxf(m, r0[j]);
    sRowMax[tid] = m;
  }
  __syncthreads();

  // ---- exp + row sum, keep exp values in registers ------------------------
#pragma unroll
  for (int vr = 0; vr < 8; ++vr) {
    const int M = vr + 8 * hi;
    const float mx = sRowMax[M];
    float s = 0.f;
#pragma unroll
    for (int t = 0; t < 8; ++t) {
      const float e = __expf(acc[t][vr] - mx);
      acc[t][vr] = e;
      s += e;
    }
    sRed[M * 128 + wv * 16 + lane15] = s;
  }
  __syncthreads();
  if (tid < 16) {
    float s = 0.f;
    const float* r0 = sRed + tid * 128;
    for (int j = 0; j < 128; ++j) s += r0[j];
    sRowSum[tid] = 1.0f / s;                 // reciprocal, survives phase 2
  }
  __syncthreads();

  // ---- phase 2: conv weights -> LDS, zero padded h1 buffer ----------------
  if (tid < 72)                 sCW[tid]            = c1w[tid];
  if (tid >= 72 && tid < 80)    sCW[tid]            = c1b[tid - 72];
  if (tid >= 128 && tid < 200)  sCW[80 + tid - 128] = c2w[tid - 128];
  if (tid == 255)               sCW[152]            = c2b[0];
  for (int i = tid; i < 8 * 34 * 34; i += 256) sH1[i] = 0.f;  // borders stay 0
  __syncthreads();

  // ---- per-query-row: attention image -> conv1(ReLU) -> conv2 -> residual -
  for (int r = 0; r < 16; ++r) {
    __syncthreads();   // protect sImg/sH1 from previous iteration's readers
    // scatter normalized attention row r (1024 values) into sImg
    if (hi == (r >> 3)) {
      const float inv = sRowSum[r];
#pragma unroll
      for (int t = 0; t < 8; ++t) {
        const int N = ((wv * 8 + t) << 4) + lane15;
        sImg[N] = v8_get(acc[t], r & 7) * inv;
      }
    }
    __syncthreads();

    // conv1: 1 -> 8 channels, 3x3 SAME, ReLU; write into padded 34x34 LDS
#pragma unroll
    for (int pi = 0; pi < 4; ++pi) {
      const int p = tid + pi * 256;
      const int y = p >> 5, x = p & 31;
      float nb[9];
#pragma unroll
      for (int dy = 0; dy < 3; ++dy)
#pragma unroll
        for (int dx = 0; dx < 3; ++dx) {
          const int yy = y + dy - 1, xx = x + dx - 1;
          nb[dy * 3 + dx] =
              (yy >= 0 && yy < 32 && xx >= 0 && xx < 32) ? sImg[yy * 32 + xx] : 0.f;
        }
#pragma unroll
      for (int ch = 0; ch < 8; ++ch) {
        float a = sCW[72 + ch];
#pragma unroll
        for (int k = 0; k < 9; ++k) a = fmaf(sCW[ch * 9 + k], nb[k], a);
        sH1[ch * 1156 + (y + 1) * 34 + (x + 1)] = fmaxf(a, 0.f);
      }
    }
    __syncthreads();

    // conv2: 8 -> 1 channel, 3x3 SAME + residual, store to global
    const size_t outBase = ((size_t)(b * NQ + qt * 16 + r)) * (size_t)NHW;
#pragma unroll
    for (int pi = 0; pi < 4; ++pi) {
      const int p = tid + pi * 256;
      const int y = p >> 5, x = p & 31;
      float o = sCW[152];
#pragma unroll
      for (int ch = 0; ch < 8; ++ch)
#pragma unroll
        for (int dy = 0; dy < 3; ++dy)
#pragma unroll
          for (int dx = 0; dx < 3; ++dx)
            o = fmaf(sCW[80 + ch * 9 + dy * 3 + dx],
                     sH1[ch * 1156 + (y + dy) * 34 + (x + dx)], o);
      out[outBase + p] = sImg[p] + o;
    }
  }
}

extern "C" void kernel_launch(void* const* d_in, const int* in_sizes, int n_in,
                              void* d_out, int out_size, void* d_ws, size_t ws_size,
                              hipStream_t stream) {
  (void)in_sizes; (void)n_in; (void)out_size; (void)d_ws; (void)ws_size;
  const float* queries  = (const float*)d_in[0];
  const float* keys     = (const float*)d_in[1];
  const int*   pos      = (const int*)d_in[2];
  const float* rel_bias = (const float*)d_in[3];
  const float* c1w      = (const float*)d_in[4];
  const float* c1b      = (const float*)d_in[5];
  const float* c2w      = (const float*)d_in[6];
  const float* c2b      = (const float*)d_in[7];
  float* out = (float*)d_out;

  dim3 grid(NB * (NQ / 16));   // 1024 blocks: (batch, 16-query tile)
  dim3 block(256);             // 8 wave32 waves
  fused_grid_attn_conv<<<grid, block, 0, stream>>>(
      queries, keys, pos, rel_bias, c1w, c1b, c2w, c2b, out);
}
